// NonEquivariantDecoderBlock_25881472925813
// MI455X (gfx1250) — compile-verified, matrix operand order checked by reading, not run on hardware
//
#include <hip/hip_runtime.h>
#include <hip/hip_bf16.h>
#include <math.h>

#define NN   1536
#define DD   256
#define KK   32
#define KS_  32
#define PP   128
#define HH   8
#define SS   64
#define NB_  16
#define RC_  32

typedef __attribute__((ext_vector_type(16))) _Float16 v16h;
typedef __attribute__((ext_vector_type(8)))  _Float16 v8h;
typedef __attribute__((ext_vector_type(8)))  float    v8f;

__device__ __forceinline__ float geluf(float x) {
  float x3 = x * x * x;
  return 0.5f * x * (1.f + tanhf(0.7978845608028654f * (x + 0.044715f * x3)));
}

__device__ __forceinline__ v16h cat8(v8h lo, v8h hi) {
  return __builtin_shufflevector(lo, hi, 0, 1, 2, 3, 4, 5, 6, 7,
                                 8, 9, 10, 11, 12, 13, 14, 15);
}

// ---------------------------------------------------------------------------
// Fragment packing: convert f32 matrices into WMMA-fragment-ordered f16.
// A layout: [rowtile][kchunk][lane]{16h}; B layout: [coltile][kchunk][lane]{16h}
// ---------------------------------------------------------------------------
__global__ void pack_a(const float* __restrict__ A, _Float16* __restrict__ Ap,
                       int nrows, int kdim) {
  int idx = blockIdx.x * blockDim.x + threadIdx.x;
  int nk = kdim >> 5;
  int total = (nrows >> 4) * nk * 32;
  if (idx >= total) return;
  int lane = idx & 31;
  int ck = (idx >> 5) % nk;
  int tr = (idx >> 5) / nk;
  int m = lane & 15, kh = (lane >> 4) * 8;
  const float* Ar = A + (size_t)(tr * 16 + m) * kdim + ck * 32;
  v16h v;
  #pragma unroll
  for (int t = 0; t < 8; ++t) {
    v[t]     = (_Float16)Ar[kh + t];
    v[t + 8] = (_Float16)Ar[16 + kh + t];
  }
  ((v16h*)Ap)[idx] = v;
}

__global__ void pack_b(const float* __restrict__ B, _Float16* __restrict__ Bp,
                       int kdim, int mcols) {
  int idx = blockIdx.x * blockDim.x + threadIdx.x;
  int nk = kdim >> 5;
  int total = (mcols >> 4) * nk * 32;
  if (idx >= total) return;
  int lane = idx & 31;
  int ck = (idx >> 5) % nk;
  int tc = (idx >> 5) / nk;
  int m = lane & 15, kh = (lane >> 4) * 8;
  const float* Bc = B + (size_t)(ck * 32) * mcols + tc * 16 + m;
  v16h v;
  #pragma unroll
  for (int t = 0; t < 8; ++t) {
    v[t]     = (_Float16)Bc[(size_t)(kh + t) * mcols];
    v[t + 8] = (_Float16)Bc[(size_t)(16 + kh + t) * mcols];
  }
  ((v16h*)Bp)[idx] = v;
}

// ---------------------------------------------------------------------------
// Packed GEMM: C = A @ B [+bias][+resid]; one wave per 16x16 tile.
// Inner loop: 2x 32B coalesced fragment loads + 1 WMMA per K-chunk.
// ---------------------------------------------------------------------------
template <bool BIAS, bool RESID>
__global__ void gemm16p(const _Float16* __restrict__ Ap, const _Float16* __restrict__ Bp,
                        float* __restrict__ C, int kdim, int mcols,
                        const float* __restrict__ bias, const float* __restrict__ resid) {
  int tilesM = mcols >> 4;
  int tile = blockIdx.x;
  int tr = tile / tilesM, tc = tile % tilesM;
  int lane = threadIdx.x & 31;
  int m = lane & 15, g = lane >> 4;
  int nk = kdim >> 5;
  const v16h* Af = (const v16h*)Ap + (size_t)tr * nk * 32 + lane;
  const v16h* Bf = (const v16h*)Bp + (size_t)tc * nk * 32 + lane;
  v8f acc = {};
  for (int ck = 0; ck < nk; ++ck) {
    acc = __builtin_amdgcn_wmma_f32_16x16x32_f16(false, Af[ck * 32], false, Bf[ck * 32],
                                                 (short)0, acc, false, false);
  }
  int col = tc * 16 + m;
  float bv = BIAS ? bias[col] : 0.f;
  #pragma unroll
  for (int r = 0; r < 8; ++r) {
    int row = tr * 16 + r + g * 8;
    float v = acc[r] + bv;
    if (RESID) v += resid[(size_t)row * mcols + col];
    C[(size_t)row * mcols + col] = v;
  }
}

// ---------------------------------------------------------------------------
// Row LayerNorm: one wave per row.
// ---------------------------------------------------------------------------
__global__ void ln_rows(const float* __restrict__ X, const float* __restrict__ s,
                        const float* __restrict__ b, float* __restrict__ Y, int ncols) {
  int row = blockIdx.x;
  int lane = threadIdx.x;
  const float* xr = X + (size_t)row * ncols;
  float sum = 0.f, sq = 0.f;
  for (int c = lane; c < ncols; c += 32) { float v = xr[c]; sum += v; sq += v * v; }
  for (int off = 16; off; off >>= 1) {
    sum += __shfl_xor(sum, off, 32);
    sq  += __shfl_xor(sq,  off, 32);
  }
  float mu = sum / (float)ncols;
  float inv = rsqrtf(sq / (float)ncols - mu * mu + 1e-5f);
  float* yr = Y + (size_t)row * ncols;
  for (int c = lane; c < ncols; c += 32) yr[c] = (xr[c] - mu) * inv * s[c] + b[c];
}

// ---------------------------------------------------------------------------
// Fused pairwise kernel: dcode -> LN -> 32x64 gelu MLP -> 64x16 -> log_softmax
// -> dmap expectation -> gumbel distance. Packed f16 weights.
// grid = (N, N/64), block = 128 (4 waves, each one 16-wide j tile).
// ---------------------------------------------------------------------------
__global__ void pair_distmap(const float* __restrict__ dl, const float* __restrict__ dr,
                             const float* __restrict__ q_rel,
                             const float* __restrict__ lns, const float* __restrict__ lnb,
                             const _Float16* __restrict__ w1p, const float* __restrict__ b1v,
                             const _Float16* __restrict__ w2p, const float* __restrict__ b2v,
                             const int* __restrict__ resi, const int* __restrict__ chain,
                             const int* __restrict__ batch, const float* __restrict__ mask,
                             const float* __restrict__ gum,
                             float* __restrict__ dmap, float* __restrict__ dist) {
  __shared__ __align__(16) _Float16 hsh[4][16][64];
  int i = blockIdx.x;
  int w = threadIdx.x >> 5;
  int lane = threadIdx.x & 31;
  int j0 = (blockIdx.y * 4 + w) * 16;
  int m = lane & 15, g = lane >> 4, kh = g * 8;
  int j = j0 + m;

  int ri = resi[i], ci = chain[i], bi = batch[i];
  int dlt = resi[j] - ri;
  dlt = dlt < -RC_ ? -RC_ : (dlt > RC_ ? RC_ : dlt);
  int ridx = dlt + RC_;
  bool sc = (chain[j] == ci) && (batch[j] == bi);

  float cv[16];
  float sum = 0.f, sq = 0.f;
  #pragma unroll
  for (int t = 0; t < 16; ++t) {
    int ch = (t < 8) ? (kh + t) : (16 + kh + (t - 8));
    float v = dl[i * 32 + ch] + dr[j * 32 + ch] + (sc ? q_rel[ridx * 32 + ch] : 0.f);
    cv[t] = v; sum += v; sq += v * v;
  }
  sum += __shfl_xor(sum, 16, 32);
  sq  += __shfl_xor(sq, 16, 32);
  float mu = sum * (1.f / 32.f);
  float inv = rsqrtf(sq * (1.f / 32.f) - mu * mu + 1e-5f);
  v16h a;
  #pragma unroll
  for (int t = 0; t < 16; ++t) {
    int ch = (t < 8) ? (kh + t) : (16 + kh + (t - 8));
    a[t] = (_Float16)((cv[t] - mu) * inv * lns[ch] + lnb[ch]);
  }

  // layer 1: (16x32)@(32x64) = 4 WMMAs (packed B, kchunks=1 per coltile)
  #pragma unroll
  for (int blk = 0; blk < 4; ++blk) {
    v16h b = ((const v16h*)w1p)[blk * 32 + lane];
    v8f hc = {};
    hc = __builtin_amdgcn_wmma_f32_16x16x32_f16(false, a, false, b, (short)0, hc,
                                                false, false);
    int col = blk * 16 + m;
    float bb = b1v[col];
    #pragma unroll
    for (int r = 0; r < 8; ++r)
      hsh[w][r + g * 8][col] = (_Float16)geluf(hc[r] + bb);
  }
  __syncthreads();

  // layer 2: (16x64)@(64x16) = 2 chained WMMAs
  const _Float16* hr = &hsh[w][m][0];
  v16h a0 = cat8(*(const v8h*)(hr + kh),      *(const v8h*)(hr + 16 + kh));
  v16h a1 = cat8(*(const v8h*)(hr + 32 + kh), *(const v8h*)(hr + 48 + kh));
  v16h b0  = ((const v16h*)w2p)[lane];
  v16h b1f = ((const v16h*)w2p)[32 + lane];
  v8f lc = {};
  lc = __builtin_amdgcn_wmma_f32_16x16x32_f16(false, a0, false, b0, (short)0, lc,
                                              false, false);
  lc = __builtin_amdgcn_wmma_f32_16x16x32_f16(false, a1, false, b1f, (short)0, lc,
                                              false, false);
  float bcls = b2v[m];
  const float step = 22.f / (float)NB_;
  float center = ((float)m + 0.5f) * step;
  #pragma unroll
  for (int r = 0; r < 8; ++r) {
    float v = lc[r] + bcls;
    float mx = v;
    for (int off = 8; off; off >>= 1) mx = fmaxf(mx, __shfl_xor(mx, off, 32));
    float e = __expf(v - mx);
    float se = e;
    for (int off = 8; off; off >>= 1) se += __shfl_xor(se, off, 32);
    float lg = v - mx - __logf(se);
    float t = __expf(lg) * center;
    for (int off = 8; off; off >>= 1) t += __shfl_xor(t, off, 32);
    if (m == 0) {
      int jj = j0 + r + g * 8;
      size_t idx = (size_t)i * NN + jj;
      dmap[idx] = t;
      float gu = gum[idx];
      bool si = (batch[jj] == bi) && (mask[i] > 0.f) && (mask[jj] > 0.f);
      dist[idx] = si ? (3.f * __logf(t + 1e-6f) + __logf(-__logf(gu)))
                     : __builtin_inff();
    }
  }
}

// ---------------------------------------------------------------------------
// Iterative top-K (K=32, ascending distance) per row.
// ---------------------------------------------------------------------------
__global__ void topk_rows(const float* __restrict__ dist, int* __restrict__ nb) {
  __shared__ float vals[NN];
  __shared__ float rmin[128];
  __shared__ int   ridx[128];
  int i = blockIdx.x;
  int t = threadIdx.x;
  for (int c = t; c < NN; c += 128) vals[c] = dist[(size_t)i * NN + c];
  __syncthreads();
  for (int sel = 0; sel < KK; ++sel) {
    float bv = __builtin_inff();
    int bi = -1;
    for (int c = t; c < NN; c += 128) {
      float v = vals[c];
      if (v < bv) { bv = v; bi = c; }
    }
    rmin[t] = bv; ridx[t] = bi;
    __syncthreads();
    for (int off = 64; off; off >>= 1) {
      if (t < off && rmin[t + off] < rmin[t]) { rmin[t] = rmin[t + off]; ridx[t] = ridx[t + off]; }
      __syncthreads();
    }
    if (t == 0) {
      int wsel = ridx[0];
      nb[i * KK + sel] = (isinf(rmin[0]) || wsel < 0) ? -1 : wsel;
      if (wsel >= 0) vals[wsel] = __builtin_inff();
    }
    __syncthreads();
  }
}

// ---------------------------------------------------------------------------
// distogram_logits for the N*KS sampled pairs (tiny FLOPs: VALU).
// ---------------------------------------------------------------------------
__global__ void distogram(const float* __restrict__ dl, const float* __restrict__ dr,
                          const float* __restrict__ q_rel,
                          const float* __restrict__ lns, const float* __restrict__ lnb,
                          const float* __restrict__ w1, const float* __restrict__ b1v,
                          const float* __restrict__ w2, const float* __restrict__ b2v,
                          const int* __restrict__ resi, const int* __restrict__ chain,
                          const int* __restrict__ batch, const int* __restrict__ sup,
                          float* __restrict__ out) {
  int idx = blockIdx.x * blockDim.x + threadIdx.x;
  if (idx >= NN * KS_) return;
  int i = idx / KS_;
  int j = sup[idx];
  int dlt = resi[j] - resi[i];
  dlt = dlt < -RC_ ? -RC_ : (dlt > RC_ ? RC_ : dlt);
  int ridx = dlt + RC_;
  bool sc = (chain[j] == chain[i]) && (batch[j] == batch[i]);
  float nc[32];
  float sum = 0.f, sq = 0.f;
  for (int ch = 0; ch < 32; ++ch) {
    float v = dl[i * 32 + ch] + dr[j * 32 + ch] + (sc ? q_rel[ridx * 32 + ch] : 0.f);
    nc[ch] = v; sum += v; sq += v * v;
  }
  float mu = sum * (1.f / 32.f);
  float inv = rsqrtf(sq * (1.f / 32.f) - mu * mu + 1e-5f);
  for (int ch = 0; ch < 32; ++ch) nc[ch] = (nc[ch] - mu) * inv * lns[ch] + lnb[ch];
  float lg[NB_];
  for (int c = 0; c < NB_; ++c) lg[c] = b2v[c];
  for (int o = 0; o < 64; ++o) {
    float acc = b1v[o];
    for (int ch = 0; ch < 32; ++ch) acc += nc[ch] * w1[ch * 64 + o];
    acc = geluf(acc);
    for (int c = 0; c < NB_; ++c) lg[c] += acc * w2[o * 16 + c];
  }
  float mx = lg[0];
  for (int c = 1; c < NB_; ++c) mx = fmaxf(mx, lg[c]);
  float se = 0.f;
  for (int c = 0; c < NB_; ++c) se += __expf(lg[c] - mx);
  float ls = __logf(se);
  float* op = out + (size_t)idx * NB_;
  for (int c = 0; c < NB_; ++c) op[c] = lg[c] - mx - ls;
}

// ---------------------------------------------------------------------------
// Fused pair-stack: pair features -> LN(P) -> 128->256 gelu ->128 -> pair@a_b.
// Packed f16 weights; one wave per 16 rows; block = 64 (2 waves).
// ---------------------------------------------------------------------------
__global__ void pair_mlp(const float* __restrict__ p_rel, const float* __restrict__ p_rbf,
                         const float* __restrict__ x1l1, const float* __restrict__ x1l2,
                         const float* __restrict__ plns, const float* __restrict__ plnb,
                         const _Float16* __restrict__ w1p, const float* __restrict__ b1v,
                         const _Float16* __restrict__ w2p, const float* __restrict__ b2v,
                         const float* __restrict__ a_bw, const int* __restrict__ nb,
                         const float* __restrict__ dmap, const int* __restrict__ resi,
                         const int* __restrict__ chain, const float* __restrict__ mask,
                         float* __restrict__ attbias) {
  __shared__ __align__(16) _Float16 lnsh[2][16][128];
  __shared__ __align__(16) _Float16 hsh[2][16][256];
  int w = threadIdx.x >> 5, lane = threadIdx.x & 31;
  int r0 = (blockIdx.x * 2 + w) * 16;
  int m = lane & 15, g = lane >> 4, kh = g * 8;
  int row = r0 + m;
  int i = row >> 5;              // KK == 32
  int jn = nb[row];
  int jj = jn < 0 ? (NN - 1) : jn;
  float pm = mask[i] * mask[jj] * (jn >= 0 ? 1.f : 0.f);
  int dlt = resi[jj] - resi[i];
  dlt = dlt < -RC_ ? -RC_ : (dlt > RC_ ? RC_ : dlt);
  int ridx = dlt + RC_;
  bool scn = (chain[jj] == chain[i]) && (jn >= 0);
  float dn = dmap[(size_t)i * NN + jj];

  const float step = 22.f / (float)NB_;
  float rb[NB_];
  #pragma unroll
  for (int bb = 0; bb < NB_; ++bb) {
    float z = (dn - ((float)bb + 0.5f) * step) / step;
    rb[bb] = (pm > 0.f) ? __expf(-0.5f * z * z) : 0.f;
  }

  float cvv[64];
  float sum = 0.f, sq = 0.f;
  for (int t = 0; t < 64; ++t) {
    int ch = g * 64 + t;
    float v = x1l1[(size_t)i * PP + ch] + x1l2[(size_t)jj * PP + ch];
    if (scn) v += p_rel[ridx * PP + ch];
    float acc = 0.f;
    #pragma unroll
    for (int bb = 0; bb < NB_; ++bb) acc += rb[bb] * p_rbf[bb * PP + ch];
    v += acc;
    cvv[t] = v; sum += v; sq += v * v;
  }
  sum += __shfl_xor(sum, 16, 32);
  sq  += __shfl_xor(sq, 16, 32);
  float mu = sum * (1.f / 128.f);
  float inv = rsqrtf(sq * (1.f / 128.f) - mu * mu + 1e-5f);
  for (int t = 0; t < 64; ++t) {
    int ch = g * 64 + t;
    lnsh[w][m][ch] = (_Float16)((cvv[t] - mu) * inv * plns[ch] + plnb[ch]);
  }
  __syncthreads();

  // layer 1: (16x128)@(128x256), 64 WMMAs
  const _Float16* lrow = &lnsh[w][m][0];
  for (int blk = 0; blk < 16; ++blk) {
    v8f acc = {};
    for (int ck = 0; ck < 4; ++ck) {
      int kb = ck * 32;
      v16h a = cat8(*(const v8h*)(lrow + kb + kh), *(const v8h*)(lrow + kb + 16 + kh));
      v16h b = ((const v16h*)w1p)[(blk * 4 + ck) * 32 + lane];
      acc = __builtin_amdgcn_wmma_f32_16x16x32_f16(false, a, false, b, (short)0, acc,
                                                   false, false);
    }
    int col = blk * 16 + m;
    float bb = b1v[col];
    #pragma unroll
    for (int r = 0; r < 8; ++r)
      hsh[w][r + g * 8][col] = (_Float16)geluf(acc[r] + bb);
  }
  __syncthreads();

  // layer 2: (16x256)@(256x128), 64 WMMAs; result back into lnsh
  const _Float16* hrow = &hsh[w][m][0];
  for (int blk = 0; blk < 8; ++blk) {
    v8f acc = {};
    for (int ck = 0; ck < 8; ++ck) {
      int kb = ck * 32;
      v16h a = cat8(*(const v8h*)(hrow + kb + kh), *(const v8h*)(hrow + kb + 16 + kh));
      v16h b = ((const v16h*)w2p)[(blk * 8 + ck) * 32 + lane];
      acc = __builtin_amdgcn_wmma_f32_16x16x32_f16(false, a, false, b, (short)0, acc,
                                                   false, false);
    }
    int col = blk * 16 + m;
    float bb = b2v[col];
    #pragma unroll
    for (int r = 0; r < 8; ++r)
      lnsh[w][r + g * 8][col] = (_Float16)(acc[r] + bb);
  }
  __syncthreads();

  // attention bias: (16 rows x 8 heads), 128 length-128 dots
  for (int qq = 0; qq < 4; ++qq) {
    int t = lane + 32 * qq;
    int rr = t >> 3, hh = t & 7;
    float acc = 0.f;
    for (int ch = 0; ch < PP; ++ch)
      acc += (float)lnsh[w][rr][ch] * a_bw[ch * HH + hh];
    attbias[(size_t)(r0 + rr) * HH + hh] = acc;
  }
}

// ---------------------------------------------------------------------------
// Neighbor attention: one block per i, wave h = head h, lane = neighbor k.
// ---------------------------------------------------------------------------
__global__ void attn(const float* __restrict__ qw, const float* __restrict__ kw,
                     const float* __restrict__ vw, const float* __restrict__ attb,
                     const int* __restrict__ nb, const float* __restrict__ mask,
                     float* __restrict__ o) {
  __shared__ float ash[HH][KK];
  __shared__ int jsh[KK];
  int i = blockIdx.x;
  int h = threadIdx.x >> 5, lane = threadIdx.x & 31;
  int jn = nb[i * KK + lane];
  int jj = jn < 0 ? (NN - 1) : jn;
  float pm = mask[i] * mask[jj] * (jn >= 0 ? 1.f : 0.f);
  if (h == 0) jsh[lane] = jj;
  const float* qv = qw + (size_t)i * (HH * SS) + h * SS;
  const float* kv = kw + (size_t)jj * (HH * SS) + h * SS;
  float dot = 0.f;
  for (int s = 0; s < SS; ++s) dot += qv[s] * kv[s];
  float av = (pm > 0.f) ? dot * 0.125f + attb[(size_t)(i * KK + lane) * HH + h] : -1e9f;
  float mx = av;
  for (int off = 16; off; off >>= 1) mx = fmaxf(mx, __shfl_xor(mx, off, 32));
  float e = __expf(av - mx);
  float se = e;
  for (int off = 16; off; off >>= 1) se += __shfl_xor(se, off, 32);
  ash[h][lane] = e / se;
  __syncthreads();
  for (int s = lane; s < SS; s += 32) {
    float acc = 0.f;
    for (int k2 = 0; k2 < KK; ++k2)
      acc += ash[h][k2] * vw[(size_t)jsh[k2] * (HH * SS) + h * SS + s];
    o[(size_t)i * (HH * SS) + h * SS + s] = acc;
  }
}

// ---------------------------------------------------------------------------
// Transition gating (in place) + segment means + hidden combine
// ---------------------------------------------------------------------------
__global__ void gatek(const float* __restrict__ u, float* __restrict__ lg,
                      float* __restrict__ cg, float* __restrict__ bg, int n) {
  int idx = blockIdx.x * blockDim.x + threadIdx.x;
  if (idx >= n) return;
  float uv = u[idx];
  lg[idx] = geluf(lg[idx]) * uv;
  cg[idx] = geluf(cg[idx]) * uv;
  bg[idx] = geluf(bg[idx]) * uv;
}

__global__ void segmean(const float* __restrict__ x, const float* __restrict__ mask,
                        float* __restrict__ out, int nseg, int rows_per, int cols) {
  int idx = blockIdx.x * blockDim.x + threadIdx.x;
  if (idx >= nseg * cols) return;
  int s = idx / cols, c = idx % cols;
  float num = 0.f, den = 0.f;
  for (int r = 0; r < rows_per; ++r) {
    int row = s * rows_per + r;
    num += x[(size_t)row * cols + c] * mask[row];
    den += mask[row];
  }
  out[idx] = num / fmaxf(den, 1.f);
}

__global__ void hiddenk(const float* __restrict__ bmean, const float* __restrict__ cmean,
                        const float* __restrict__ lgu, const int* __restrict__ batch,
                        const int* __restrict__ chain, float* __restrict__ hid) {
  int idx = blockIdx.x * blockDim.x + threadIdx.x;
  if (idx >= NN * DD * 2) return;
  int n = idx / (DD * 2), c = idx % (DD * 2);
  hid[idx] = bmean[batch[n] * (DD * 2) + c] + cmean[chain[n] * (DD * 2) + c] + lgu[idx];
}

// ---------------------------------------------------------------------------
extern "C" void kernel_launch(void* const* d_in, const int* in_sizes, int n_in,
                              void* d_out, int out_size, void* d_ws, size_t ws_size,
                              hipStream_t stream) {
  (void)in_sizes; (void)n_in; (void)out_size; (void)ws_size;
  const float* features = (const float*)d_in[0];
  const float* q_dl   = (const float*)d_in[1];
  const float* q_dr   = (const float*)d_in[2];
  const float* q_rel  = (const float*)d_in[3];
  const float* q_ln_s = (const float*)d_in[4];
  const float* q_ln_b = (const float*)d_in[5];
  const float* q_m1   = (const float*)d_in[6];
  const float* q_m1b  = (const float*)d_in[7];
  const float* q_m2   = (const float*)d_in[8];
  const float* q_m2b  = (const float*)d_in[9];
  const float* ln1_s  = (const float*)d_in[10];
  const float* ln1_b  = (const float*)d_in[11];
  const float* p_rel  = (const float*)d_in[12];
  const float* p_rbf  = (const float*)d_in[13];
  const float* p_l1   = (const float*)d_in[14];
  const float* p_l2   = (const float*)d_in[15];
  const float* p_ln_s = (const float*)d_in[16];
  const float* p_ln_b = (const float*)d_in[17];
  const float* p_m1   = (const float*)d_in[18];
  const float* p_m1b  = (const float*)d_in[19];
  const float* p_m2   = (const float*)d_in[20];
  const float* p_m2b  = (const float*)d_in[21];
  const float* ln2_s  = (const float*)d_in[22];
  const float* ln2_b  = (const float*)d_in[23];
  const float* a_q    = (const float*)d_in[24];
  const float* a_k    = (const float*)d_in[25];
  const float* a_v    = (const float*)d_in[26];
  const float* a_b    = (const float*)d_in[27];
  const float* a_o    = (const float*)d_in[28];
  const float* a_ob   = (const float*)d_in[29];
  const float* ln3_s  = (const float*)d_in[30];
  const float* ln3_b  = (const float*)d_in[31];
  const float* u_u    = (const float*)d_in[32];
  const float* u_lg   = (const float*)d_in[33];
  const float* u_cg   = (const float*)d_in[34];
  const float* u_bg   = (const float*)d_in[35];
  const float* u_o    = (const float*)d_in[36];
  const float* u_ob   = (const float*)d_in[37];
  const float* gumbel = (const float*)d_in[38];
  const int*   resi   = (const int*)d_in[39];
  const int*   chain  = (const int*)d_in[40];
  const int*   batch  = (const int*)d_in[41];
  const float* mask   = (const float*)d_in[42];
  const int*   sup    = (const int*)d_in[43];

  float* outFeat = (float*)d_out;
  float* outDg   = (float*)d_out + (size_t)NN * DD;

  size_t off = 0;
  #define WSF(name, cnt) \
    off = (off + 63) & ~(size_t)63; \
    float* name = (float*)((char*)d_ws + off); off += (size_t)(cnt) * sizeof(float);
  #define WSH(name, cnt) \
    off = (off + 63) & ~(size_t)63; \
    _Float16* name = (_Float16*)((char*)d_ws + off); off += (size_t)(cnt) * sizeof(_Float16);
  WSF(dmap,  (size_t)NN * NN)
  WSF(dist,  (size_t)NN * NN)
  WSF(dl,    NN * 32)
  WSF(dr,    NN * 32)
  WSF(x1,    NN * DD)
  WSF(x1l1,  NN * PP)
  WSF(x1l2,  NN * PP)
  WSF(x2,    NN * DD)
  WSF(qw,    NN * HH * SS)
  WSF(kw,    NN * HH * SS)
  WSF(vw,    NN * HH * SS)
  WSF(attb,  NN * KK * HH)
  WSF(ows,   NN * HH * SS)
  WSF(feat2, NN * DD)
  WSF(x3,    NN * DD)
  WSF(uws,   NN * DD * 2)
  WSF(lgws,  NN * DD * 2)
  WSF(cgws,  NN * DD * 2)
  WSF(bgws,  NN * DD * 2)
  WSF(bmean, 3 * DD * 2)
  WSF(cmean, 6 * DD * 2)
  WSF(hid,   NN * DD * 2)
  // packed f16 activations
  WSH(featP, NN * DD)
  WSH(x1P,   NN * DD)
  WSH(x2P,   NN * DD)
  WSH(x3P,   NN * DD)
  WSH(owsP,  NN * HH * SS)
  WSH(hidP,  NN * DD * 2)
  // packed f16 weights
  WSH(qdlP,  DD * 32)
  WSH(qdrP,  DD * 32)
  WSH(pl1P,  DD * PP)
  WSH(pl2P,  DD * PP)
  WSH(aqP,   DD * HH * SS)
  WSH(akP,   DD * HH * SS)
  WSH(avP,   DD * HH * SS)
  WSH(aoP,   HH * SS * DD)
  WSH(uuP,   DD * DD * 2)
  WSH(ulgP,  DD * DD * 2)
  WSH(ucgP,  DD * DD * 2)
  WSH(ubgP,  DD * DD * 2)
  WSH(uoP,   DD * 2 * DD)
  WSH(qm1P,  32 * 64)
  WSH(qm2P,  64 * 16)
  WSH(pm1P,  PP * 256)
  WSH(pm2P,  256 * PP)
  off = (off + 63) & ~(size_t)63;
  int* nb = (int*)((char*)d_ws + off); off += (size_t)NN * KK * sizeof(int);
  #undef WSF
  #undef WSH

  #define PACKA(src, dst, nrows, kdim) \
    pack_a<<<dim3((((nrows) >> 4) * ((kdim) >> 5) * 32 + 255) / 256), dim3(256), 0, stream>>>( \
        src, dst, nrows, kdim);
  #define PACKB(src, dst, kdim, mcols) \
    pack_b<<<dim3((((mcols) >> 4) * ((kdim) >> 5) * 32 + 255) / 256), dim3(256), 0, stream>>>( \
        src, dst, kdim, mcols);

  // ---- stage 1: dl/dr projections --------------------------------------
  PACKA(features, featP, NN, DD)
  PACKB(q_dl, qdlP, DD, 32)
  PACKB(q_dr, qdrP, DD, 32)
  gemm16p<false, false><<<dim3((NN / 16) * (32 / 16)), dim3(32), 0, stream>>>(
      featP, qdlP, dl, DD, 32, nullptr, nullptr);
  gemm16p<false, false><<<dim3((NN / 16) * (32 / 16)), dim3(32), 0, stream>>>(
      featP, qdrP, dr, DD, 32, nullptr, nullptr);

  // ---- stage 2: fused pairwise distogram -> dmap / dist ----------------
  PACKB(q_m1, qm1P, 32, 64)
  PACKB(q_m2, qm2P, 64, 16)
  pair_distmap<<<dim3(NN, NN / 64), dim3(128), 0, stream>>>(
      dl, dr, q_rel, q_ln_s, q_ln_b, qm1P, q_m1b, qm2P, q_m2b,
      resi, chain, batch, mask, gumbel, dmap, dist);

  // ---- stage 3: top-K + sampled distogram logits -----------------------
  topk_rows<<<dim3(NN), dim3(128), 0, stream>>>(dist, nb);
  distogram<<<dim3((NN * KS_ + 255) / 256), dim3(256), 0, stream>>>(
      dl, dr, q_rel, q_ln_s, q_ln_b, q_m1, q_m1b, q_m2, q_m2b,
      resi, chain, batch, sup, outDg);

  // ---- stage 4: pair stack --------------------------------------------
  ln_rows<<<dim3(NN), dim3(32), 0, stream>>>(features, ln1_s, ln1_b, x1, DD);
  PACKA(x1, x1P, NN, DD)
  PACKB(p_l1, pl1P, DD, PP)
  PACKB(p_l2, pl2P, DD, PP)
  gemm16p<false, false><<<dim3((NN / 16) * (PP / 16)), dim3(32), 0, stream>>>(
      x1P, pl1P, x1l1, DD, PP, nullptr, nullptr);
  gemm16p<false, false><<<dim3((NN / 16) * (PP / 16)), dim3(32), 0, stream>>>(
      x1P, pl2P, x1l2, DD, PP, nullptr, nullptr);
  PACKB(p_m1, pm1P, PP, 256)
  PACKB(p_m2, pm2P, 256, PP)
  pair_mlp<<<dim3((NN * KK) / 32), dim3(64), 0, stream>>>(
      p_rel, p_rbf, x1l1, x1l2, p_ln_s, p_ln_b, pm1P, p_m1b, pm2P, p_m2b,
      a_b, nb, dmap, resi, chain, mask, attb);

  // ---- stage 5: attention ---------------------------------------------
  ln_rows<<<dim3(NN), dim3(32), 0, stream>>>(features, ln2_s, ln2_b, x2, DD);
  PACKA(x2, x2P, NN, DD)
  PACKB(a_q, aqP, DD, HH * SS)
  PACKB(a_k, akP, DD, HH * SS)
  PACKB(a_v, avP, DD, HH * SS)
  gemm16p<false, false><<<dim3((NN / 16) * (HH * SS / 16)), dim3(32), 0, stream>>>(
      x2P, aqP, qw, DD, HH * SS, nullptr, nullptr);
  gemm16p<false, false><<<dim3((NN / 16) * (HH * SS / 16)), dim3(32), 0, stream>>>(
      x2P, akP, kw, DD, HH * SS, nullptr, nullptr);
  gemm16p<false, false><<<dim3((NN / 16) * (HH * SS / 16)), dim3(32), 0, stream>>>(
      x2P, avP, vw, DD, HH * SS, nullptr, nullptr);
  attn<<<dim3(NN), dim3(256), 0, stream>>>(qw, kw, vw, attb, nb, mask, ows);
  PACKA(ows, owsP, NN, HH * SS)
  PACKB(a_o, aoP, HH * SS, DD)
  gemm16p<true, true><<<dim3((NN / 16) * (DD / 16)), dim3(32), 0, stream>>>(
      owsP, aoP, feat2, HH * SS, DD, a_ob, features);

  // ---- stage 6: transition --------------------------------------------
  ln_rows<<<dim3(NN), dim3(32), 0, stream>>>(feat2, ln3_s, ln3_b, x3, DD);
  PACKA(x3, x3P, NN, DD)
  PACKB(u_u,  uuP,  DD, DD * 2)
  PACKB(u_lg, ulgP, DD, DD * 2)
  PACKB(u_cg, ucgP, DD, DD * 2)
  PACKB(u_bg, ubgP, DD, DD * 2)
  gemm16p<false, false><<<dim3((NN / 16) * (DD * 2 / 16)), dim3(32), 0, stream>>>(
      x3P, uuP, uws, DD, DD * 2, nullptr, nullptr);
  gemm16p<false, false><<<dim3((NN / 16) * (DD * 2 / 16)), dim3(32), 0, stream>>>(
      x3P, ulgP, lgws, DD, DD * 2, nullptr, nullptr);
  gemm16p<false, false><<<dim3((NN / 16) * (DD * 2 / 16)), dim3(32), 0, stream>>>(
      x3P, ucgP, cgws, DD, DD * 2, nullptr, nullptr);
  gemm16p<false, false><<<dim3((NN / 16) * (DD * 2 / 16)), dim3(32), 0, stream>>>(
      x3P, ubgP, bgws, DD, DD * 2, nullptr, nullptr);

  int ntot = NN * DD * 2;
  gatek<<<dim3((ntot + 255) / 256), dim3(256), 0, stream>>>(uws, lgws, cgws, bgws, ntot);
  segmean<<<dim3((3 * DD * 2 + 255) / 256), dim3(256), 0, stream>>>(
      bgws, mask, bmean, 3, NN / 3, DD * 2);
  segmean<<<dim3((6 * DD * 2 + 255) / 256), dim3(256), 0, stream>>>(
      cgws, mask, cmean, 6, NN / 6, DD * 2);
  hiddenk<<<dim3((ntot + 255) / 256), dim3(256), 0, stream>>>(
      bmean, cmean, lgws, batch, chain, hid);

  PACKA(hid, hidP, NN, DD * 2)
  PACKB(u_o, uoP, DD * 2, DD)
  gemm16p<true, true><<<dim3((NN / 16) * (DD / 16)), dim3(32), 0, stream>>>(
      hidP, uoP, outFeat, DD * 2, DD, u_ob, feat2);

  #undef PACKA
  #undef PACKB
}
